// SeqUGA_11287174054106
// MI455X (gfx1250) — matrix operand order repaired and checked by reading, not compiled, not run on hardware
//
#include <hip/hip_runtime.h>
#include <hip/hip_bf16.h>
#include <math.h>

// ---------------------------------------------------------------------------
// CDNA5 (gfx1250) implementation of the 3-layer graph transformer.
// Big GEMMs: v_wmma_f32_16x16x32_f16, 64x256 block tile, 32x64 wave tile
// (8 WMMA / wave / K-chunk), double-buffered LDS with packed f16 staging
// (v_cvt_pk_rtz_f16_f32 + b32 DS stores).
// ---------------------------------------------------------------------------

typedef __attribute__((ext_vector_type(16))) _Float16 v16h;
typedef __attribute__((ext_vector_type(2)))  __fp16   v2hf;   // cvt_pkrtz type
typedef __attribute__((ext_vector_type(8)))  float    v8f;

#define LDSS 34   // 32 halfs + 2 pad

union FragU { v16h h; unsigned int u[8]; };
union PackU { v2hf h2; unsigned int u; };

// ---------------------------------------------------------------------------
// C[M,N] = act(A[M,K] @ W[K,N] + bias[N]); act = exact GELU if gelu != 0.
// block = 256 threads (8 wave32). Block tile 64x256; wave (wm,wn) -> 32x64.
// Requires M%64==0, N%256==0, K%32==0 (holds: M=40000, N in {256,1024}).
// ---------------------------------------------------------------------------
__global__ __launch_bounds__(256) void gemm_wmma_f16_kernel(
    const float* __restrict__ A, const float* __restrict__ W,
    const float* __restrict__ bias, float* __restrict__ C,
    int M, int N, int K, int gelu)
{
  __shared__ _Float16 As[2][64 * LDSS];     // [buf][m][k]
  __shared__ _Float16 Bs[2][256 * LDSS];    // [buf][n][k] (transposed)

  const int tid  = threadIdx.x;
  const int wave = tid >> 5;
  const int lane = tid & 31;
  const int wm   = wave & 1;      // m sub-tile: 0..1 (32 rows)
  const int wn   = wave >> 1;     // n sub-tile: 0..3 (64 cols)
  const int m0   = blockIdx.x * 64;
  const int n0   = blockIdx.y * 256;
  const int l16  = lane & 15;
  const int hi   = lane >> 4;     // 0 | 1

  // staging mapping: thread owns K-PAIRS so f16 packing -> single b32 store
  const int ar  = tid >> 4;          // A row base 0..15 (rows ar+16i)
  const int ac2 = (tid & 15) * 2;    // A col pair 0,2,...,30
  const int bn  = tid;               // B column 0..255 (16 k-pairs each)

  float2 aP[4];
  float2 bP[16];

  auto loadChunk = [&](int k0) {
    #pragma unroll
    for (int i = 0; i < 4; ++i)
      aP[i] = *(const float2*)&A[(long)(m0 + ar + 16 * i) * K + (k0 + ac2)];
    #pragma unroll
    for (int j = 0; j < 16; ++j) {
      bP[j].x = W[(long)(k0 + 2 * j)     * N + (n0 + bn)];
      bP[j].y = W[(long)(k0 + 2 * j + 1) * N + (n0 + bn)];
    }
  };
  auto storeChunk = [&](int buf) {
    #pragma unroll
    for (int i = 0; i < 4; ++i) {
      PackU p; p.h2 = __builtin_amdgcn_cvt_pkrtz(aP[i].x, aP[i].y);
      *(unsigned int*)&As[buf][(ar + 16 * i) * LDSS + ac2] = p.u;
    }
    #pragma unroll
    for (int j = 0; j < 16; ++j) {
      PackU p; p.h2 = __builtin_amdgcn_cvt_pkrtz(bP[j].x, bP[j].y);
      *(unsigned int*)&Bs[buf][bn * LDSS + 2 * j] = p.u;
    }
  };

  v8f c0[4] = {{}, {}, {}, {}};   // m-tile 0, n-tiles 0..3
  v8f c1[4] = {{}, {}, {}, {}};   // m-tile 1, n-tiles 0..3

  loadChunk(0);
  storeChunk(0);
  int buf = 0;
  const int nCh = K >> 5;

  for (int ch = 0; ch < nCh; ++ch) {
    __syncthreads();
    if (ch + 1 < nCh) loadChunk((ch + 1) << 5);   // overlap with WMMA below

    // ---- fragments per ISA 7.12.2 (16-bit layouts)
    FragU a0, a1, b[4];
    {
      const _Float16* as = As[buf];
      int m  = 32 * wm + l16;
      int kb = hi * 8;
      #pragma unroll
      for (int j = 0; j < 4; ++j) {
        a0.u[j]     = *(const unsigned int*)&as[m * LDSS + kb + 2 * j];
        a0.u[4 + j] = *(const unsigned int*)&as[m * LDSS + 16 + kb + 2 * j];
        a1.u[j]     = *(const unsigned int*)&as[(m + 16) * LDSS + kb + 2 * j];
        a1.u[4 + j] = *(const unsigned int*)&as[(m + 16) * LDSS + 16 + kb + 2 * j];
      }
    }
    {
      const _Float16* bs = Bs[buf];
      int nb = 64 * wn + l16;
      int kb = hi * 8;
      #pragma unroll
      for (int t = 0; t < 4; ++t)
        #pragma unroll
        for (int j = 0; j < 8; ++j)
          b[t].u[j] = *(const unsigned int*)&bs[(nb + 16 * t) * LDSS + kb + 2 * j + (j >= 4 ? 8 : 0)];
    }

    #pragma unroll
    for (int t = 0; t < 4; ++t) {
      c0[t] = __builtin_amdgcn_wmma_f32_16x16x32_f16(false, a0.h, false, b[t].h,
                                                     (short)0, c0[t], false, false);
      c1[t] = __builtin_amdgcn_wmma_f32_16x16x32_f16(false, a1.h, false, b[t].h,
                                                     (short)0, c1[t], false, false);
    }

    if (ch + 1 < nCh) storeChunk(buf ^ 1);        // fill alternate buffer
    buf ^= 1;
  }

  // ---- epilogue: bias (+ exact GELU); C/D layout m = r + 8*hi, n = lane&15
  const int mb = m0 + 32 * wm + hi * 8;
  const int nb = n0 + 64 * wn + l16;
  #pragma unroll
  for (int t = 0; t < 4; ++t) {
    int n = nb + 16 * t;
    float bsv = bias[n];
    #pragma unroll
    for (int r = 0; r < 8; ++r) {
      float v0 = c0[t][r] + bsv;
      float v1 = c1[t][r] + bsv;
      if (gelu) {
        v0 = 0.5f * v0 * (1.0f + erff(v0 * 0.70710678118654752f));
        v1 = 0.5f * v1 * (1.0f + erff(v1 * 0.70710678118654752f));
      }
      C[(long)(mb + r) * N + n]      = v0;
      C[(long)(mb + 16 + r) * N + n] = v1;
    }
  }
}

// ---------------------------------------------------------------------------
// x[row,:] = emb[items[row],:]   (D = 256 -> 64 float4 per row)
// ---------------------------------------------------------------------------
__global__ void embed_gather_kernel(const int* __restrict__ items,
                                    const float* __restrict__ emb,
                                    float* __restrict__ x, int D)
{
  long row = blockIdx.x;
  const float4* s = (const float4*)(emb + (long)items[row] * D);
  float4*       d = (float4*)(x + row * D);
  d[threadIdx.x] = s[threadIdx.x];
}

// ---------------------------------------------------------------------------
// scores[(b*H+h)*E + e] = dot(q[b,src[e],h,:], k[b,dst[e],h,:])
// ---------------------------------------------------------------------------
__global__ void edge_scores_kernel(const float* __restrict__ q,
                                   const float* __restrict__ k,
                                   const int* __restrict__ ei,
                                   float* __restrict__ scores,
                                   int Bn, int N, int E, int H, int HD)
{
  long idx = (long)blockIdx.x * 256 + threadIdx.x;
  long total = (long)Bn * E * H;
  if (idx >= total) return;
  int h = (int)(idx % H);
  long r = idx / H;
  int e = (int)(r % E);
  int b = (int)(r / E);
  int s = ei[e], d = ei[E + e];
  const float4* qp = (const float4*)(q + ((long)(b * N + s) * H + h) * HD);
  const float4* kp = (const float4*)(k + ((long)(b * N + d) * H + h) * HD);
  float acc = 0.f;
  #pragma unroll
  for (int i = 0; i < 16; ++i) {   // HD = 64
    float4 a = qp[i], c = kp[i];
    acc += a.x * c.x + a.y * c.y + a.z * c.z + a.w * c.w;
  }
  scores[(long)(b * H + h) * E + e] = acc;
}

// ---------------------------------------------------------------------------
// per (b,h): max & sum(exp) over all E edges (softmax over axis=1 of (B,E,H))
// ---------------------------------------------------------------------------
__global__ __launch_bounds__(256) void softmax_reduce_kernel(
    const float* __restrict__ scores, float* __restrict__ smax,
    float* __restrict__ ssum, int E)
{
  __shared__ float red[256];
  const float* s = scores + (long)blockIdx.x * E;
  int t = threadIdx.x;
  float m = -1e30f;
  for (int i = t; i < E; i += 256) m = fmaxf(m, s[i]);
  red[t] = m; __syncthreads();
  for (int o = 128; o > 0; o >>= 1) {
    if (t < o) red[t] = fmaxf(red[t], red[t + o]);
    __syncthreads();
  }
  float mx = red[0]; __syncthreads();
  float acc = 0.f;
  for (int i = t; i < E; i += 256) acc += expf(s[i] - mx);
  red[t] = acc; __syncthreads();
  for (int o = 128; o > 0; o >>= 1) {
    if (t < o) red[t] += red[t + o];
    __syncthreads();
  }
  if (t == 0) { smax[blockIdx.x] = mx; ssum[blockIdx.x] = red[0]; }
}

__global__ void zero_kernel(float* __restrict__ p, long n)
{
  long i = (long)blockIdx.x * blockDim.x + threadIdx.x;
  if (i < n) p[i] = 0.f;
}

// ---------------------------------------------------------------------------
// agg[b,dst,h,:] += softmax_w * v[b,src,h,:]   one block per (b,e), 256 thr = D
// ---------------------------------------------------------------------------
__global__ __launch_bounds__(256) void scatter_agg_kernel(
    const float* __restrict__ v, const float* __restrict__ scores,
    const float* __restrict__ smax, const float* __restrict__ ssum,
    const int* __restrict__ ei, float* __restrict__ agg,
    int N, int E, int H, int HD)
{
  int e = blockIdx.x % E;
  int b = blockIdx.x / E;
  int t = threadIdx.x;              // t = h*HD + hd = d
  int h = t / HD;
  int bh = b * H + h;
  float w = expf(scores[(long)bh * E + e] - smax[bh]) / ssum[bh];
  int s = ei[e], d = ei[E + e];
  int D = H * HD;
  float val = w * v[(long)(b * N + s) * D + t];
  atomicAdd(&agg[(long)(b * N + d) * D + t], val);
}

// ---------------------------------------------------------------------------
// x = LayerNorm(x + attn) ; one block (256 thr) per row, D = 256
// ---------------------------------------------------------------------------
__global__ __launch_bounds__(256) void add_ln_kernel(
    float* __restrict__ x, const float* __restrict__ attn,
    const float* __restrict__ g, const float* __restrict__ bta, int D)
{
  __shared__ float red[256];
  long row = blockIdx.x;
  int t = threadIdx.x;
  float v = x[row * D + t] + attn[row * D + t];
  red[t] = v; __syncthreads();
  for (int o = 128; o > 0; o >>= 1) { if (t < o) red[t] += red[t + o]; __syncthreads(); }
  float mean = red[0] / (float)D; __syncthreads();
  float dv = v - mean;
  red[t] = dv * dv; __syncthreads();
  for (int o = 128; o > 0; o >>= 1) { if (t < o) red[t] += red[t + o]; __syncthreads(); }
  float var = red[0] / (float)D;
  x[row * D + t] = dv * rsqrtf(var + 1e-5f) * g[t] + bta[t];
}

// pooled[b,d] = mean_n x[b,n,d]
__global__ void pool_mean_kernel(const float* __restrict__ x,
                                 float* __restrict__ pooled, int N, int D)
{
  int idx = blockIdx.x * blockDim.x + threadIdx.x; // b*D + d
  int d = idx % D, b = idx / D;
  const float* p = x + (long)b * N * D + d;
  float acc = 0.f;
  for (int n = 0; n < N; ++n) acc += p[(long)n * D];
  pooled[idx] = acc / (float)N;
}

// out[b,v] = pooled[b,:] . Wout[:,v] + bout[v]
// one thread per vocab column; streams Wout exactly once (51 MB total),
// accumulating all B=8 outputs; pooled staged in LDS.
__global__ __launch_bounds__(256) void vocab_proj_kernel(
    const float* __restrict__ pooled, const float* __restrict__ Wout,
    const float* __restrict__ bout, float* __restrict__ out, int D, int V)
{
  __shared__ float sp[8 * 256];     // B * D
  for (int i = threadIdx.x; i < 8 * D; i += 256) sp[i] = pooled[i];
  __syncthreads();

  int v = blockIdx.x * 256 + threadIdx.x;
  if (v >= V) return;
  float acc[8];
  float bv = bout[v];
  #pragma unroll
  for (int b = 0; b < 8; ++b) acc[b] = bv;
  for (int d = 0; d < D; ++d) {
    float w = Wout[(long)d * V + v];
    #pragma unroll
    for (int b = 0; b < 8; ++b) acc[b] += sp[b * D + d] * w;
  }
  #pragma unroll
  for (int b = 0; b < 8; ++b) out[(long)b * V + v] = acc[b];
}

// ---------------------------------------------------------------------------
extern "C" void kernel_launch(void* const* d_in, const int* in_sizes, int n_in,
                              void* d_out, int out_size, void* d_ws, size_t ws_size,
                              hipStream_t stream)
{
  const int*   items = (const int*)d_in[0];
  const int*   ei    = (const int*)d_in[1];
  const float* emb   = (const float*)d_in[2];
  const float* Wq = (const float*)d_in[3];  const float* bq = (const float*)d_in[4];
  const float* Wk = (const float*)d_in[5];  const float* bk = (const float*)d_in[6];
  const float* Wv = (const float*)d_in[7];  const float* bv = (const float*)d_in[8];
  const float* Wo = (const float*)d_in[9];  const float* bo = (const float*)d_in[10];
  const float* W1 = (const float*)d_in[11]; const float* b1 = (const float*)d_in[12];
  const float* W2 = (const float*)d_in[13]; const float* b2 = (const float*)d_in[14];
  const float* lng = (const float*)d_in[15]; const float* lnb = (const float*)d_in[16];
  const float* Wout = (const float*)d_in[17]; const float* bout = (const float*)d_in[18];
  float* out = (float*)d_out;

  const int B   = 8;                         // reference constant
  const int V   = in_sizes[18];              // 50000
  const int D   = in_sizes[2] / V;           // 256
  const int L   = in_sizes[3] / (D * D);     // 3
  const int H   = 4;                         // reference constant
  const int HD  = D / H;                     // 64
  const int FF  = in_sizes[12] / L;          // 1024
  const int M   = in_sizes[0];               // B*SEQ = 40000
  const int N   = M / B;                     // SEQ = 5000
  const int E   = in_sizes[1] / 2;           // 9800

  const long MD = (long)M * D;

  // workspace carve (floats)
  float* ws     = (float*)d_ws;
  float* xbuf   = ws;                        // MD
  float* attn   = ws + MD;                   // MD
  float* S      = ws + 2 * MD;               // 4*MD  (q,k,v,agg | ff hidden)
  float* qb     = S;
  float* kb     = S + MD;
  float* vb     = S + 2 * MD;
  float* agg    = S + 3 * MD;
  float* h1     = S;                         // reuses q/k/v/agg region (M x FF)
  float* scores = ws + 6 * MD;               // B*H*E
  float* smax   = scores + (long)B * H * E;  // B*H
  float* ssum   = smax + B * H;              // B*H
  float* pooled = ssum + B * H;              // B*D

  // ---- embedding gather
  embed_gather_kernel<<<M, D / 4, 0, stream>>>(items, emb, xbuf, D);

  for (int l = 0; l < L; ++l) {
    const float* Wq_l = Wq + (long)l * D * D;  const float* bq_l = bq + (long)l * D;
    const float* Wk_l = Wk + (long)l * D * D;  const float* bk_l = bk + (long)l * D;
    const float* Wv_l = Wv + (long)l * D * D;  const float* bv_l = bv + (long)l * D;
    const float* Wo_l = Wo + (long)l * D * D;  const float* bo_l = bo + (long)l * D;
    const float* W1_l = W1 + (long)l * D * FF; const float* b1_l = b1 + (long)l * FF;
    const float* W2_l = W2 + (long)l * FF * D; const float* b2_l = b2 + (long)l * D;

    dim3 gDD(M / 64, D / 256);                 // N = 256 -> y = 1
    gemm_wmma_f16_kernel<<<gDD, 256, 0, stream>>>(xbuf, Wq_l, bq_l, qb, M, D, D, 0);
    gemm_wmma_f16_kernel<<<gDD, 256, 0, stream>>>(xbuf, Wk_l, bk_l, kb, M, D, D, 0);
    gemm_wmma_f16_kernel<<<gDD, 256, 0, stream>>>(xbuf, Wv_l, bv_l, vb, M, D, D, 0);

    long nScore = (long)B * E * H;
    edge_scores_kernel<<<(unsigned)((nScore + 255) / 256), 256, 0, stream>>>(
        qb, kb, ei, scores, B, N, E, H, HD);
    softmax_reduce_kernel<<<B * H, 256, 0, stream>>>(scores, smax, ssum, E);

    zero_kernel<<<(unsigned)((MD + 255) / 256), 256, 0, stream>>>(agg, MD);
    scatter_agg_kernel<<<B * E, H * HD, 0, stream>>>(vb, scores, smax, ssum, ei,
                                                     agg, N, E, H, HD);

    gemm_wmma_f16_kernel<<<gDD, 256, 0, stream>>>(agg, Wo_l, bo_l, attn, M, D, D, 0);

    dim3 gDF(M / 64, FF / 256);                // N = 1024 -> y = 4
    gemm_wmma_f16_kernel<<<gDF, 256, 0, stream>>>(attn, W1_l, b1_l, h1, M, FF, D, 1);
    gemm_wmma_f16_kernel<<<gDD, 256, 0, stream>>>(h1, W2_l, b2_l, xbuf, M, D, FF, 0);

    add_ln_kernel<<<M, D, 0, stream>>>(xbuf, attn, lng + (long)l * D,
                                       lnb + (long)l * D, D);
  }

  pool_mean_kernel<<<(B * D) / 256, 256, 0, stream>>>(xbuf, pooled, N, D);
  vocab_proj_kernel<<<(unsigned)((V + 255) / 256), 256, 0, stream>>>(
      pooled, Wout, bout, out, D, V);
}
